// LinearGLUMoELayer_45655502356776
// MI455X (gfx1250) — matrix operand order repaired and checked by reading, not compiled, not run on hardware
//
#include <hip/hip_runtime.h>
#include <hip/hip_bf16.h>

typedef __attribute__((ext_vector_type(16))) __bf16 v16bf;
typedef __attribute__((ext_vector_type(2)))  __bf16 v2bf;
typedef __attribute__((ext_vector_type(8)))  float  v8f;

#define T_TOK 4096
#define DDIM  2048
#define HE    1376
#define ODIM  2048
#define EE    8
#define ROWS_CAP 8704   // 2*T padded up to 63 per expert, 64-aligned
#define NGROUPS  136    // ROWS_CAP / 64

// ---- packed f32x2 -> bf16x2 conversion (single VALU op) ----
#if __has_builtin(__builtin_amdgcn_cvt_pk_bf16_f32)
__device__ __forceinline__ unsigned int cvt_pk_bf16(float a, float b) {
    union { v2bf v; unsigned int u; } r;
    r.v = __builtin_amdgcn_cvt_pk_bf16_f32(a, b);
    return r.u;
}
#else
__device__ __forceinline__ unsigned int cvt_pk_bf16(float a, float b) {
    unsigned int r;
    asm volatile("v_cvt_pk_bf16_f32 %0, %1, %2" : "=v"(r) : "v"(a), "v"(b));
    return r;
}
#endif

__device__ __forceinline__ unsigned short f2bf(float f) {
    return (unsigned short)(cvt_pk_bf16(f, 0.f) & 0xFFFFu);
}

union AU { v16bf v; uint4 q[2]; };
union BU { v16bf v; unsigned int w[8]; };

// load 16 contiguous f32, convert to bf16 vector via packed converts
__device__ __forceinline__ void ldw16_bf(const float* p, BU& b) {
    const float4* q = (const float4*)p;
    float4 f0 = q[0], f1 = q[1], f2 = q[2], f3 = q[3];
    b.w[0] = cvt_pk_bf16(f0.x, f0.y);
    b.w[1] = cvt_pk_bf16(f0.z, f0.w);
    b.w[2] = cvt_pk_bf16(f1.x, f1.y);
    b.w[3] = cvt_pk_bf16(f1.z, f1.w);
    b.w[4] = cvt_pk_bf16(f2.x, f2.y);
    b.w[5] = cvt_pk_bf16(f2.z, f2.w);
    b.w[6] = cvt_pk_bf16(f3.x, f3.y);
    b.w[7] = cvt_pk_bf16(f3.z, f3.w);
}

__device__ __forceinline__ float softplus_f(float v) {
    return v > 20.f ? v : log1pf(expf(v));
}

// ---------------- init: routing state ----------------
__global__ void init_kernel(int* counts, int* cursors, int* group_expert,
                            int* tok_ids, float* tok_sc) {
    int i = blockIdx.x * 256 + threadIdx.x;
    if (i < EE) { counts[i] = 0; cursors[i] = 0; }
    if (i < NGROUPS) group_expert[i] = -1;
    if (i < ROWS_CAP) { tok_ids[i] = 0; tok_sc[i] = 0.f; }
}

// ---------------- prep: zero y + convert x -> bf16 (packed) ----------------
__global__ void prep_kernel(const float* __restrict__ x, unsigned int* __restrict__ xb32,
                            float2* __restrict__ y2, long long npairs) {
    long long i = (long long)blockIdx.x * blockDim.x + threadIdx.x;
    long long stride = (long long)gridDim.x * blockDim.x;
    for (; i < npairs; i += stride) {
        float a = x[2 * i], b = x[2 * i + 1];
        xb32[i] = cvt_pk_bf16(a, b);
        y2[i] = make_float2(0.f, 0.f);      // npairs == T*D/2 == T*O/2
    }
}

// ---------------- gating: one wave per token ----------------
__global__ __launch_bounds__(256) void gating_kernel(
    const float* __restrict__ x, const float* __restrict__ noise,
    const float* __restrict__ wg1, const float* __restrict__ wg2,
    const float* __restrict__ wn,
    int* __restrict__ top_idx, float* __restrict__ top_sc,
    float* __restrict__ imp_tok, float* __restrict__ load_tok,
    int* __restrict__ counts)
{
    int wave = threadIdx.x >> 5;
    int lane = threadIdx.x & 31;
    int t = blockIdx.x * 8 + wave;
    if (t >= T_TOK) return;
    const float* xr = x + (size_t)t * DDIM;

    float acc[16];
#pragma unroll
    for (int r = 0; r < 16; ++r) acc[r] = 0.f;
    for (int d = lane; d < DDIM; d += 32) {
        float xv = xr[d];
#pragma unroll
        for (int r = 0; r < 8; ++r) acc[r]     += xv * wg1[r * DDIM + d];
#pragma unroll
        for (int r = 0; r < 8; ++r) acc[8 + r] += xv * wn[r * DDIM + d];
    }
#pragma unroll
    for (int r = 0; r < 16; ++r)
        for (int off = 16; off > 0; off >>= 1)
            acc[r] += __shfl_xor(acc[r], off, 32);

    if (lane != 0) return;

    float a[EE], lg[EE], nc[EE], lt[EE];
#pragma unroll
    for (int e = 0; e < EE; ++e) a[e] = tanhf(acc[e]);
#pragma unroll
    for (int e = 0; e < EE; ++e) {
        float s = 0.f;
#pragma unroll
        for (int j = 0; j < EE; ++j) s += a[j] * wg2[e * EE + j];
        lg[e] = s;
    }
#pragma unroll
    for (int e = 0; e < EE; ++e) nc[e] = softplus_f(acc[8 + e]) + 0.01f;
#pragma unroll
    for (int e = 0; e < EE; ++e) lt[e] = lg[e] + nc[e] * noise[(size_t)t * EE + e];

    // top-3 of 8
    int i0 = 0, i1 = 0, i2 = 0;
    float v0 = -1e30f, v1 = -1e30f, v2 = -1e30f;
#pragma unroll
    for (int e = 0; e < EE; ++e) if (lt[e] > v0) { v0 = lt[e]; i0 = e; }
#pragma unroll
    for (int e = 0; e < EE; ++e) if (e != i0 && lt[e] > v1) { v1 = lt[e]; i1 = e; }
#pragma unroll
    for (int e = 0; e < EE; ++e) if (e != i0 && e != i1 && lt[e] > v2) { v2 = lt[e]; i2 = e; }
    (void)i2;

    float e1 = expf(v1 - v0);
    float inv = 1.f / (1.f + e1);
    float sc0 = inv, sc1 = e1 * inv;

    top_idx[t * 2] = i0;  top_idx[t * 2 + 1] = i1;
    top_sc[t * 2]  = sc0; top_sc[t * 2 + 1]  = sc1;
    atomicAdd(&counts[i0], 1);
    atomicAdd(&counts[i1], 1);

    float thr_in = v2, thr_out = v1;
#pragma unroll
    for (int e = 0; e < EE; ++e) {
        float im = (e == i0) ? sc0 : ((e == i1) ? sc1 : 0.f);
        imp_tok[(size_t)t * EE + e] = im;
        bool isin = lt[e] > thr_in;
        float thr = isin ? thr_in : thr_out;
        float z = (lg[e] - thr) / nc[e];
        load_tok[(size_t)t * EE + e] = 0.5f * (1.0f + erff(z * 0.70710678118f));
    }
}

// ---------------- offsets: 64-aligned prefix sum + group->expert map ----------------
__global__ void offsets_kernel(const int* __restrict__ counts,
                               int* __restrict__ group_expert, int* __restrict__ offsets) {
    if (threadIdx.x != 0 || blockIdx.x != 0) return;
    int off = 0;
    for (int e = 0; e < EE; ++e) {
        offsets[e] = off;
        int ng = (counts[e] + 63) >> 6;
        for (int j = 0; j < ng; ++j) group_expert[(off >> 6) + j] = e;
        off += ng * 64;
    }
    offsets[EE] = off;
}

// ---------------- scatter tokens into expert buckets ----------------
__global__ void scatter_kernel(const int* __restrict__ top_idx, const float* __restrict__ top_sc,
                               const int* __restrict__ offsets, int* __restrict__ cursors,
                               int* __restrict__ tok_ids, float* __restrict__ tok_sc) {
    int t = blockIdx.x * 256 + threadIdx.x;
    if (t >= T_TOK) return;
#pragma unroll
    for (int i = 0; i < 2; ++i) {
        int e = top_idx[t * 2 + i];
        int pos = atomicAdd(&cursors[e], 1);
        int slot = offsets[e] + pos;
        tok_ids[slot] = t;
        tok_sc[slot]  = top_sc[t * 2 + i];
    }
}

// ---------------- GEMM1: 64 tokens x 16 h per wave, bf16 WMMA ----------------
__global__ __launch_bounds__(32) void expert_gemm1(
    const float* __restrict__ wgp, const float* __restrict__ wup,
    const float* __restrict__ bgp, const float* __restrict__ bup,
    const unsigned short* __restrict__ xb,
    const int* __restrict__ group_expert, const int* __restrict__ tok_ids,
    unsigned short* __restrict__ hb)
{
    int grp = blockIdx.x;
    int e = group_expert[grp];
    if (e < 0) return;
    int h0 = blockIdx.y * 16;
    int lane = threadIdx.x, hi = lane >> 4, n = lane & 15;
    int base = grp * 64;

    const unsigned short* arp[4];
#pragma unroll
    for (int g = 0; g < 4; ++g) {
        int tok = tok_ids[base + g * 16 + n];              // A row M = lane&15
        arp[g] = xb + (size_t)tok * DDIM;
    }
    const float* wgr = wgp + ((size_t)e * HE + h0 + n) * DDIM;
    const float* wur = wup + ((size_t)e * HE + h0 + n) * DDIM;

    v8f cg[4], cu[4];
#pragma unroll
    for (int g = 0; g < 4; ++g)
#pragma unroll
        for (int i = 0; i < 8; ++i) { cg[g][i] = 0.f; cu[g][i] = 0.f; }

    for (int kk = 0; kk < DDIM / 32; ++kk) {
        int k0 = kk * 32;
        AU a[4];
#pragma unroll
        for (int g = 0; g < 4; ++g) {
            a[g].q[0] = *(const uint4*)(arp[g] + k0 + hi * 16);      // K = k0+hi*16..+7
            a[g].q[1] = *(const uint4*)(arp[g] + k0 + hi * 16 + 8);  // K = k0+hi*16+8..+15
        }
        BU bg_, bu_;
        ldw16_bf(wgr + k0 + hi * 16, bg_);                 // col n, 16 contiguous K
        ldw16_bf(wur + k0 + hi * 16, bu_);
#pragma unroll
        for (int g = 0; g < 4; ++g) {
            cg[g] = __builtin_amdgcn_wmma_f32_16x16x32_bf16(false, a[g].v, false, bg_.v, (short)0, cg[g], false, false);
            cu[g] = __builtin_amdgcn_wmma_f32_16x16x32_bf16(false, a[g].v, false, bu_.v, (short)0, cu[g], false, false);
        }
    }

    float bgv = bgp[(size_t)e * HE + h0 + n];
    float buv = bup[(size_t)e * HE + h0 + n];
#pragma unroll
    for (int g = 0; g < 4; ++g) {
#pragma unroll
        for (int r = 0; r < 8; ++r) {
            int m = g * 16 + hi * 8 + r;                   // C: vgpr r -> M=r (+8 hi lanes)
            float gv = cg[g][r] + bgv;
            float uv = cu[g][r] + buv;
            float hv = (gv / (1.f + expf(-gv))) * uv;      // silu(g)*u
            hb[(size_t)(base + m) * HE + h0 + n] = f2bf(hv);
        }
    }
}

// ---------------- GEMM2: 64 tokens x 16 o per wave, bf16 WMMA ----------------
__global__ __launch_bounds__(32) void expert_gemm2(
    const float* __restrict__ wdp, const float* __restrict__ bdp,
    const unsigned short* __restrict__ hb,
    const int* __restrict__ group_expert, const int* __restrict__ tok_ids,
    const float* __restrict__ tok_sc, float* __restrict__ y)
{
    int grp = blockIdx.x;
    int e = group_expert[grp];
    if (e < 0) return;
    int o0 = blockIdx.y * 16;
    int lane = threadIdx.x, hi = lane >> 4, n = lane & 15;
    int base = grp * 64;

    const unsigned short* arp[4];
#pragma unroll
    for (int g = 0; g < 4; ++g)
        arp[g] = hb + (size_t)(base + g * 16 + n) * HE;
    const float* wr = wdp + ((size_t)e * ODIM + o0 + n) * HE;

    v8f c[4];
#pragma unroll
    for (int g = 0; g < 4; ++g)
#pragma unroll
        for (int i = 0; i < 8; ++i) c[g][i] = 0.f;

    for (int kk = 0; kk < HE / 32; ++kk) {                 // 43 iterations
        int k0 = kk * 32;
        AU a[4];
#pragma unroll
        for (int g = 0; g < 4; ++g) {
            a[g].q[0] = *(const uint4*)(arp[g] + k0 + hi * 16);
            a[g].q[1] = *(const uint4*)(arp[g] + k0 + hi * 16 + 8);
        }
        BU b_;
        ldw16_bf(wr + k0 + hi * 16, b_);
#pragma unroll
        for (int g = 0; g < 4; ++g)
            c[g] = __builtin_amdgcn_wmma_f32_16x16x32_bf16(false, a[g].v, false, b_.v, (short)0, c[g], false, false);
    }

    float bdv = bdp[(size_t)e * ODIM + o0 + n];
#pragma unroll
    for (int g = 0; g < 4; ++g) {
#pragma unroll
        for (int r = 0; r < 8; ++r) {
            int m = g * 16 + hi * 8 + r;
            int t = tok_ids[base + m];
            float sc = tok_sc[base + m];
            atomicAdd(&y[(size_t)t * ODIM + o0 + n], sc * (c[g][r] + bdv));
        }
    }
}

// ---------------- deterministic reductions for gate loss ----------------
__global__ __launch_bounds__(256) void reduce_gate(const float* __restrict__ imp_tok,
                                                   const float* __restrict__ load_tok,
                                                   float* __restrict__ imp, float* __restrict__ loadv) {
    int e = blockIdx.x;
    int tid = threadIdx.x;
    __shared__ float s1[256], s2[256];
    float a = 0.f, b = 0.f;
    for (int t = tid; t < T_TOK; t += 256) {
        a += imp_tok[(size_t)t * EE + e];
        b += load_tok[(size_t)t * EE + e];
    }
    s1[tid] = a; s2[tid] = b;
    __syncthreads();
    for (int st = 128; st > 0; st >>= 1) {
        if (tid < st) { s1[tid] += s1[tid + st]; s2[tid] += s2[tid + st]; }
        __syncthreads();
    }
    if (tid == 0) { imp[e] = s1[0]; loadv[e] = s2[0]; }
}

__global__ void finalize_kernel(const float* __restrict__ imp, const float* __restrict__ loadv,
                                float* __restrict__ out_loss) {
    if (threadIdx.x != 0 || blockIdx.x != 0) return;
    float m1 = 0.f, m2 = 0.f;
    for (int e = 0; e < EE; ++e) { m1 += imp[e]; m2 += loadv[e]; }
    m1 *= (1.f / EE); m2 *= (1.f / EE);
    float v1 = 0.f, v2 = 0.f;
    for (int e = 0; e < EE; ++e) {
        float d1 = imp[e] - m1, d2 = loadv[e] - m2;
        v1 += d1 * d1; v2 += d2 * d2;
    }
    v1 *= (1.f / (EE - 1)); v2 *= (1.f / (EE - 1));
    float cv1 = v1 / (m1 * m1 + 1e-10f);
    float cv2 = v2 / (m2 * m2 + 1e-10f);
    out_loss[0] = (cv1 + cv2) * 0.01f;
}

extern "C" void kernel_launch(void* const* d_in, const int* in_sizes, int n_in,
                              void* d_out, int out_size, void* d_ws, size_t ws_size,
                              hipStream_t stream) {
    const float* x     = (const float*)d_in[0];
    const float* noise = (const float*)d_in[1];
    const float* wg1   = (const float*)d_in[2];
    const float* wg2   = (const float*)d_in[3];
    const float* wn    = (const float*)d_in[4];
    const float* wg    = (const float*)d_in[5];
    const float* wu    = (const float*)d_in[6];
    const float* wd    = (const float*)d_in[7];
    const float* bg    = (const float*)d_in[8];
    const float* bu    = (const float*)d_in[9];
    const float* bd    = (const float*)d_in[10];

    float* y = (float*)d_out;                    // T*O floats, then 1 gate_loss float

    // workspace layout (bytes)
    char* ws = (char*)d_ws;
    float* imp        = (float*)(ws + 0);                    // 8
    float* loadv      = (float*)(ws + 32);                   // 8
    int*   counts     = (int*)(ws + 64);                     // 8
    int*   cursors    = (int*)(ws + 96);                     // 8
    int*   offsets    = (int*)(ws + 128);                    // 9
    int*   grp_exp    = (int*)(ws + 192);                    // 136
    int*   tok_ids    = (int*)(ws + 2304);                   // 8704 -> 37120
    float* tok_sc     = (float*)(ws + 37120);                // 8704 -> 71936
    int*   top_idx    = (int*)(ws + 71936);                  // 8192 -> 104704
    float* top_sc     = (float*)(ws + 104704);               // 8192 -> 137472
    unsigned short* x_bf = (unsigned short*)(ws + 137472);   // T*D bf16 = 16 MB
    unsigned short* h_bf = (unsigned short*)(ws + 16914688); // ROWS_CAP*HE bf16 ~= 23.9 MB
    float* imp_tok    = (float*)(ws + 40868096);             // T*E
    float* load_tok   = (float*)(ws + 40999168);             // T*E (end ~41.1 MB)

    // 1. init routing state
    init_kernel<<<(ROWS_CAP + 255) / 256, 256, 0, stream>>>(counts, cursors, grp_exp, tok_ids, tok_sc);

    // 2. zero y + x -> bf16 (packed, T*D == T*O)
    prep_kernel<<<2048, 256, 0, stream>>>(x, (unsigned int*)x_bf, (float2*)y,
                                          (long long)T_TOK * DDIM / 2);

    // 3. gating (one wave per token)
    gating_kernel<<<T_TOK / 8, 256, 0, stream>>>(x, noise, wg1, wg2, wn,
                                                 top_idx, top_sc, imp_tok, load_tok, counts);

    // 4. gate loss: deterministic reduction + cv^2
    reduce_gate<<<EE, 256, 0, stream>>>(imp_tok, load_tok, imp, loadv);
    finalize_kernel<<<1, 32, 0, stream>>>(imp, loadv, y + (size_t)T_TOK * ODIM);

    // 5. bucket tokens per expert (64-row aligned segments)
    offsets_kernel<<<1, 32, 0, stream>>>(counts, grp_exp, offsets);
    scatter_kernel<<<(T_TOK + 255) / 256, 256, 0, stream>>>(top_idx, top_sc, offsets, cursors,
                                                            tok_ids, tok_sc);

    // 6. expert GEMMs (bf16 WMMA, one wave per 64x16 tile, 4x B reuse)
    dim3 g1(NGROUPS, HE / 16);    // 136 x 86
    expert_gemm1<<<g1, 32, 0, stream>>>(wg, wu, bg, bu, x_bf, grp_exp, tok_ids, h_bf);

    dim3 g2(NGROUPS, ODIM / 16);  // 136 x 128
    expert_gemm2<<<g2, 32, 0, stream>>>(wd, bd, h_bf, grp_exp, tok_ids, tok_sc, y);

    (void)in_sizes; (void)n_in; (void)out_size; (void)ws_size;
}
